// DGCNN_7713761264286
// MI455X (gfx1250) — compile-verified, matrix operand order checked by reading, not run on hardware
//
#include <hip/hip_runtime.h>
#include <hip/hip_bf16.h>

typedef __attribute__((ext_vector_type(16))) __bf16 v16bf;
typedef __attribute__((ext_vector_type(8)))  float  v8f;

#define NPTS   16384          // 8 * 2048
#define KNN_K  20
#define GY     128            // row-group grid dim -> 1024 waves per col-tile
#define EPSF   1e-5f
#define SLOPE  0.2f

// ---------- order-preserving float<->uint map for atomic max/min ----------
__device__ __forceinline__ unsigned fmap(float x) {
  unsigned u = __float_as_uint(x);
  return (u & 0x80000000u) ? ~u : (u | 0x80000000u);
}
__device__ __forceinline__ float funmap(unsigned m) {
  unsigned u = (m & 0x80000000u) ? (m & 0x7FFFFFFFu) : ~m;
  return __uint_as_float(u);
}
__device__ __forceinline__ float leaky(float y) { return y > 0.f ? y : SLOPE * y; }

// per-(point,channel) running max/min update for one D tile (rows rt*16+off+v)
__device__ __forceinline__ void mm_update(const v8f& acc, int rt, int off, int ocol, int Cout,
                                          unsigned* __restrict__ hmax,
                                          unsigned* __restrict__ hmin) {
  int v = 0;
  while (v < 8) {
    int r   = rt * 16 + off + v;
    int bn2 = r / KNN_K;
    float mx = acc[v], mn = acc[v];
    int v2 = v + 1;
    while (v2 < 8 && (rt * 16 + off + v2) / KNN_K == bn2) {
      mx = fmaxf(mx, acc[v2]); mn = fminf(mn, acc[v2]); ++v2;
    }
    size_t o = (size_t)bn2 * Cout + ocol;
    atomicMax(&hmax[o], fmap(mx));
    atomicMin(&hmin[o], fmap(mn));
    v = v2;
  }
}

// ---------------- transpose x (8,3,2048) -> f0 (16384,3) ----------------
__global__ void k_transpose(const float* __restrict__ x, float* __restrict__ f0) {
  int i = blockIdx.x * 256 + threadIdx.x;
  if (i >= NPTS * 3) return;
  int p = i / 3, c = i - p * 3;
  int b = p >> 11, n = p & 2047;
  f0[i] = x[(b * 3 + c) * 2048 + n];
}

// ---------------- KNN: one wave per point, 20x wave-argmax ----------------
__global__ __launch_bounds__(128) void k_knn(const float* __restrict__ f0,
                                             int* __restrict__ idx) {
  __shared__ float dist[4][2048];
  const int lane = threadIdx.x & 31;
  const int wave = threadIdx.x >> 5;
  const int p = blockIdx.x * 4 + wave;           // 0..16383
  const int b = p >> 11;
  const float fx = f0[p * 3 + 0], fy = f0[p * 3 + 1], fz = f0[p * 3 + 2];
  for (int m = lane; m < 2048; m += 32) {
    const float* q = f0 + (size_t)((b << 11) + m) * 3;
    float dx = q[0] - fx, dy = q[1] - fy, dz = q[2] - fz;
    dist[wave][m] = -(dx * dx + dy * dy + dz * dz);
  }
  __syncthreads();
  for (int t = 0; t < KNN_K; ++t) {
    float best = -3.4e38f;
    int bi = 1 << 30;
    for (int m = lane; m < 2048; m += 32) {
      float d = dist[wave][m];
      if (d > best || (d == best && m < bi)) { best = d; bi = m; }
    }
    for (int s = 16; s > 0; s >>= 1) {
      float ob = __shfl_xor(best, s, 32);
      int   oi = __shfl_xor(bi,   s, 32);
      if (ob > best || (ob == best && oi < bi)) { best = ob; bi = oi; }
    }
    if (lane == 0) { idx[p * KNN_K + t] = bi; dist[wave][bi] = -3.4e38f; }
    __syncthreads();
  }
}

// ---------------- init per-point max/min buffers ----------------
__global__ void k_init_mm(unsigned* __restrict__ hmax, unsigned* __restrict__ hmin, int count) {
  int i = blockIdx.x * 256 + threadIdx.x;
  if (i < count) { hmax[i] = 0u; hmin[i] = 0xFFFFFFFFu; }
}

// ---------------- EdgeConv WMMA kernel (dual accumulator chains) ----------------
// h[r,o] = sum_c feat[r,c] * W[o,c],  r = bn*20 + k, feat = [neigh-cent | cent]
template <int NKS>
__global__ __launch_bounds__(256)
void k_edge_wmma(const float* __restrict__ fin, int in_stride, int in_coloff,
                 int Chalf, int Cin, int Cout,
                 const float* __restrict__ W, const int* __restrict__ idx,
                 float* __restrict__ PS, float* __restrict__ PQ,
                 unsigned* __restrict__ hmax, unsigned* __restrict__ hmin) {
  const int lane  = threadIdx.x & 31;
  const int wave  = threadIdx.x >> 5;
  const int lr    = lane & 15;
  const int half  = lane >> 4;
  const int kbase = half * 8;
  const int off   = half * 8;
  const int ocol  = blockIdx.x * 16 + lr;

  // preload B fragments (Wt: K x N, per lane column ocol)
  v16bf Bf[NKS];
#pragma unroll
  for (int ks = 0; ks < NKS; ++ks) {
    v16bf bb;
#pragma unroll
    for (int e = 0; e < 16; ++e) {
      int c = ks * 32 + kbase + e + ((e >= 8) ? 8 : 0);
      float w = (c < Cin) ? W[(size_t)ocol * Cin + c] : 0.0f;
      bb[e] = (__bf16)w;
    }
    Bf[ks] = bb;
  }

  float s_acc = 0.f, q_acc = 0.f;
  const int NRT = (NPTS * KNN_K) / 16;  // 20480 row tiles (even)
  // each wave processes aligned tile pairs (rt, rt+1): two independent WMMA chains
  for (int rt = (blockIdx.y * 8 + wave) * 2; rt < NRT; rt += GY * 8 * 2) {
    const int gr0 = rt * 16 + lr;            // A-rows owned by this lane
    const int gr1 = gr0 + 16;
    const int bn0 = gr0 / KNN_K,  bn1 = gr1 / KNN_K;
    const int kk0 = gr0 - bn0 * KNN_K, kk1 = gr1 - bn1 * KNN_K;
    const int b0  = bn0 >> 11,    b1  = bn1 >> 11;
    const int nb0 = idx[bn0 * KNN_K + kk0];
    const int nb1 = idx[bn1 * KNN_K + kk1];
    const float* pN0 = fin + (size_t)((b0 << 11) + nb0) * in_stride + in_coloff;
    const float* pC0 = fin + (size_t)bn0 * in_stride + in_coloff;
    const float* pN1 = fin + (size_t)((b1 << 11) + nb1) * in_stride + in_coloff;
    const float* pC1 = fin + (size_t)bn1 * in_stride + in_coloff;

    v8f acc0 = {0.f, 0.f, 0.f, 0.f, 0.f, 0.f, 0.f, 0.f};
    v8f acc1 = {0.f, 0.f, 0.f, 0.f, 0.f, 0.f, 0.f, 0.f};
#pragma unroll
    for (int ks = 0; ks < NKS; ++ks) {
      v16bf a0, a1;
#pragma unroll
      for (int e = 0; e < 16; ++e) {
        int c = ks * 32 + kbase + e + ((e >= 8) ? 8 : 0);
        float v0 = 0.0f, v1 = 0.0f;
        if (c < Chalf)      { v0 = pN0[c] - pC0[c];       v1 = pN1[c] - pC1[c]; }
        else if (c < Cin)   { v0 = pC0[c - Chalf];        v1 = pC1[c - Chalf]; }
        a0[e] = (__bf16)v0;
        a1[e] = (__bf16)v1;
      }
      acc0 = __builtin_amdgcn_wmma_f32_16x16x32_bf16(false, a0, false, Bf[ks],
                                                     (short)0, acc0, false, false);
      acc1 = __builtin_amdgcn_wmma_f32_16x16x32_bf16(false, a1, false, Bf[ks],
                                                     (short)0, acc1, false, false);
    }

#pragma unroll
    for (int v = 0; v < 8; ++v) {
      float x0 = acc0[v], x1 = acc1[v];
      s_acc += x0 + x1; q_acc += x0 * x0 + x1 * x1;
    }
    mm_update(acc0, rt,     off, ocol, Cout, hmax, hmin);
    mm_update(acc1, rt + 1, off, ocol, Cout, hmax, hmin);
  }
  // deterministic per-wave partial stats (lane and lane^16 share a channel)
  s_acc += __shfl_xor(s_acc, 16, 32);
  q_acc += __shfl_xor(q_acc, 16, 32);
  if (half == 0) {
    int rg = blockIdx.y * 8 + wave;   // < 1024
    PS[rg * 512 + ocol] = s_acc;
    PQ[rg * 512 + ocol] = q_acc;
  }
}

// ---------------- embedding WMMA: h = fcat(16384x512) @ Wemb^T ----------------
__global__ __launch_bounds__(256)
void k_emb_wmma(const float* __restrict__ fcat, const float* __restrict__ W,
                float* __restrict__ hemb, float* __restrict__ PS, float* __restrict__ PQ) {
  const int lane  = threadIdx.x & 31;
  const int wave  = threadIdx.x >> 5;
  const int lr    = lane & 15;
  const int half  = lane >> 4;
  const int kbase = half * 8;
  const int off   = half * 8;
  const int ocol  = blockIdx.x * 16 + lr;

  v16bf Bf[16];
#pragma unroll
  for (int ks = 0; ks < 16; ++ks) {
    v16bf bb;
#pragma unroll
    for (int e = 0; e < 16; ++e) {
      int c = ks * 32 + kbase + e + ((e >= 8) ? 8 : 0);
      bb[e] = (__bf16)W[(size_t)ocol * 512 + c];
    }
    Bf[ks] = bb;
  }

  float s_acc = 0.f, q_acc = 0.f;
  const int NRT = NPTS / 16;  // 1024 (even)
  // waves 0..511 each process one aligned tile pair (wave-uniform bound)
  for (int rt = (blockIdx.y * 8 + wave) * 2; rt < NRT; rt += GY * 8 * 2) {
    const float* pA0 = fcat + (size_t)(rt * 16 + lr) * 512;
    const float* pA1 = pA0 + (size_t)16 * 512;
    v8f acc0 = {0.f, 0.f, 0.f, 0.f, 0.f, 0.f, 0.f, 0.f};
    v8f acc1 = {0.f, 0.f, 0.f, 0.f, 0.f, 0.f, 0.f, 0.f};
#pragma unroll
    for (int ks = 0; ks < 16; ++ks) {
      v16bf a0, a1;
#pragma unroll
      for (int e = 0; e < 16; ++e) {
        int c = ks * 32 + kbase + e + ((e >= 8) ? 8 : 0);
        a0[e] = (__bf16)pA0[c];
        a1[e] = (__bf16)pA1[c];
      }
      acc0 = __builtin_amdgcn_wmma_f32_16x16x32_bf16(false, a0, false, Bf[ks],
                                                     (short)0, acc0, false, false);
      acc1 = __builtin_amdgcn_wmma_f32_16x16x32_bf16(false, a1, false, Bf[ks],
                                                     (short)0, acc1, false, false);
    }
#pragma unroll
    for (int v = 0; v < 8; ++v) {
      float x0 = acc0[v], x1 = acc1[v];
      s_acc += x0 + x1; q_acc += x0 * x0 + x1 * x1;
      hemb[(size_t)(rt * 16 + off + v) * 512 + ocol]        = x0;
      hemb[(size_t)((rt + 1) * 16 + off + v) * 512 + ocol]  = x1;
    }
  }
  s_acc += __shfl_xor(s_acc, 16, 32);
  q_acc += __shfl_xor(q_acc, 16, 32);
  if (half == 0) {
    int rg = blockIdx.y * 8 + wave;
    PS[rg * 512 + ocol] = s_acc;
    PQ[rg * 512 + ocol] = q_acc;
  }
}

// ---------------- deterministic stats reduction ----------------
__global__ void k_stats_reduce(const float* __restrict__ PS, const float* __restrict__ PQ,
                               float* __restrict__ ssum, float* __restrict__ ssq, int Cout) {
  int c = blockIdx.x * 256 + threadIdx.x;
  if (c >= Cout) return;
  float s = 0.f, q = 0.f;
  for (int rg = 0; rg < 1024; ++rg) { s += PS[rg * 512 + c]; q += PQ[rg * 512 + c]; }
  ssum[c] = s; ssq[c] = q;
}

// ---------------- EdgeConv epilogue: BN + leaky on max/min -> fcat slice ----------------
__global__ void k_edge_apply(const unsigned* __restrict__ hmax, const unsigned* __restrict__ hmin,
                             const float* __restrict__ ssum, const float* __restrict__ ssq,
                             const float* __restrict__ ga, const float* __restrict__ be,
                             float* __restrict__ fcat, int Cout, int coloff, float invcnt) {
  int i = blockIdx.x * 256 + threadIdx.x;
  if (i >= NPTS * Cout) return;
  int bn = i / Cout, c = i - bn * Cout;
  float g = ga[c];
  float h = (g >= 0.f) ? funmap(hmax[i]) : funmap(hmin[i]);  // max commutes with monotone BN+leaky
  float mean = ssum[c] * invcnt;
  float var  = ssq[c] * invcnt - mean * mean;
  float y = g * (h - mean) * rsqrtf(var + EPSF) + be[c];
  fcat[(size_t)bn * 512 + coloff + c] = leaky(y);
}

// ---------------- emb BN+leaky + global max/mean pool -> pooled (8,1024) ----------------
__global__ void k_emb_pool(const float* __restrict__ hemb,
                           const float* __restrict__ ssum, const float* __restrict__ ssq,
                           const float* __restrict__ ga, const float* __restrict__ be,
                           float* __restrict__ pooled) {
  int t = blockIdx.x * 256 + threadIdx.x;  // 0..4095
  if (t >= 8 * 512) return;
  int b = t >> 9, o = t & 511;
  float g = ga[o], bb = be[o];
  float mean = ssum[o] * (1.f / 16384.f);
  float var  = ssq[o] * (1.f / 16384.f) - mean * mean;
  float inv  = rsqrtf(var + EPSF);
  float mx = -3.4e38f, sm = 0.f;
  const float* p = hemb + (size_t)b * 2048 * 512 + o;
  for (int n = 0; n < 2048; ++n) {
    float y = leaky(g * (p[(size_t)n * 512] - mean) * inv + bb);
    mx = fmaxf(mx, y); sm += y;
  }
  pooled[b * 1024 + o]       = mx;
  pooled[b * 1024 + 512 + o] = sm * (1.f / 2048.f);
}

// ---------------- MLP head: FC(1024->512)+BN+leaky, FC(512->256)+BN+leaky, FC(256->40)+b ----
__global__ __launch_bounds__(512)
void k_head(const float* __restrict__ pooled,
            const float* __restrict__ Wl1, const float* __restrict__ gl1, const float* __restrict__ bl1,
            const float* __restrict__ Wl2, const float* __restrict__ gl2, const float* __restrict__ bl2,
            const float* __restrict__ Wf,  const float* __restrict__ bf, float* __restrict__ out) {
  __shared__ float P[8 * 1024];
  __shared__ float H1[8 * 512];
  __shared__ float H2[8 * 256];
  const int t = threadIdx.x;
  for (int i = t; i < 8 * 1024; i += 512) P[i] = pooled[i];
  __syncthreads();
  {
    const float* w = Wl1 + (size_t)t * 1024;
    for (int r = 0; r < 8; ++r) {
      float s = 0.f; const float* p = P + r * 1024;
      for (int c = 0; c < 1024; ++c) s += p[c] * w[c];
      H1[r * 512 + t] = s;
    }
  }
  __syncthreads();
  if (t < 512) {
    float m = 0.f; for (int r = 0; r < 8; ++r) m += H1[r * 512 + t];
    m *= 0.125f;
    float v = 0.f; for (int r = 0; r < 8; ++r) { float d = H1[r * 512 + t] - m; v += d * d; }
    v *= 0.125f;
    float inv = rsqrtf(v + EPSF);
    for (int r = 0; r < 8; ++r)
      H1[r * 512 + t] = leaky(gl1[t] * (H1[r * 512 + t] - m) * inv + bl1[t]);
  }
  __syncthreads();
  if (t < 256) {
    const float* w = Wl2 + (size_t)t * 512;
    for (int r = 0; r < 8; ++r) {
      float s = 0.f; const float* p = H1 + r * 512;
      for (int c = 0; c < 512; ++c) s += p[c] * w[c];
      H2[r * 256 + t] = s;
    }
  }
  __syncthreads();
  if (t < 256) {
    float m = 0.f; for (int r = 0; r < 8; ++r) m += H2[r * 256 + t];
    m *= 0.125f;
    float v = 0.f; for (int r = 0; r < 8; ++r) { float d = H2[r * 256 + t] - m; v += d * d; }
    v *= 0.125f;
    float inv = rsqrtf(v + EPSF);
    for (int r = 0; r < 8; ++r)
      H2[r * 256 + t] = leaky(gl2[t] * (H2[r * 256 + t] - m) * inv + bl2[t]);
  }
  __syncthreads();
  if (t < 320) {
    int r = t / 40, o = t - r * 40;
    float s = bf[o];
    const float* w = Wf + (size_t)o * 256;
    const float* p = H2 + r * 256;
    for (int c = 0; c < 256; ++c) s += p[c] * w[c];
    out[t] = s;
  }
}

// =========================== launcher ===========================
extern "C" void kernel_launch(void* const* d_in, const int* in_sizes, int n_in,
                              void* d_out, int out_size, void* d_ws, size_t ws_size,
                              hipStream_t stream) {
  const float* x    = (const float*)d_in[0];
  const float* We[4] = {(const float*)d_in[1], (const float*)d_in[4], (const float*)d_in[7], (const float*)d_in[10]};
  const float* ge[4] = {(const float*)d_in[2], (const float*)d_in[5], (const float*)d_in[8], (const float*)d_in[11]};
  const float* be[4] = {(const float*)d_in[3], (const float*)d_in[6], (const float*)d_in[9], (const float*)d_in[12]};
  const float* Wemb = (const float*)d_in[13];
  const float* gemb = (const float*)d_in[14];
  const float* bemb = (const float*)d_in[15];
  const float* Wl1  = (const float*)d_in[16];
  const float* gl1  = (const float*)d_in[17];
  const float* bl1  = (const float*)d_in[18];
  const float* Wl2  = (const float*)d_in[19];
  const float* gl2  = (const float*)d_in[20];
  const float* bl2  = (const float*)d_in[21];
  const float* Wf   = (const float*)d_in[22];
  const float* bf   = (const float*)d_in[23];
  float* out = (float*)d_out;

  // ---- workspace layout (~73 MB) ----
  char* w = (char*)d_ws;
  float* f0   = (float*)w;    w += (size_t)NPTS * 3 * 4;          // 196 KB
  int*   idx  = (int*)w;      w += (size_t)NPTS * KNN_K * 4;      // 1.31 MB
  float* fcat = (float*)w;    w += (size_t)NPTS * 512 * 4;        // 33.55 MB
  unsigned* hmax = (unsigned*)w;                                  // edge scratch,
  float*    hemb = (float*)w;                                     // aliased by emb h
  unsigned* hmin = (unsigned*)(w + (size_t)NPTS * 256 * 4);
  w += (size_t)NPTS * 512 * 4;                                    // 33.55 MB
  float* PS   = (float*)w;    w += (size_t)1024 * 512 * 4;        // 2 MB
  float* PQ   = (float*)w;    w += (size_t)1024 * 512 * 4;        // 2 MB
  float* ssum = (float*)w;    w += 512 * 4;
  float* ssq  = (float*)w;    w += 512 * 4;
  float* pooled = (float*)w;  w += 8 * 1024 * 4;

  // 1) transpose + KNN (idx computed once, reused by all edge convs)
  k_transpose<<<(NPTS * 3 + 255) / 256, 256, 0, stream>>>(x, f0);
  k_knn<<<NPTS / 4, 128, 0, stream>>>(f0, idx);

  // 2) four EdgeConv layers into fcat columns [0:64 | 64:128 | 128:256 | 256:512]
  const int   Chalf[4]  = {3, 64, 64, 128};
  const int   Cin[4]    = {6, 128, 128, 256};
  const int   Cout[4]   = {64, 64, 128, 256};
  const int   incol[4]  = {0, 0, 64, 128};
  const int   outcol[4] = {0, 64, 128, 256};
  const float invcnt_e  = 1.f / (float)(NPTS * KNN_K);

  for (int l = 0; l < 4; ++l) {
    const float* fin = (l == 0) ? f0 : fcat;
    int stride = (l == 0) ? 3 : 512;
    int count = NPTS * Cout[l];
    k_init_mm<<<(count + 255) / 256, 256, 0, stream>>>(hmax, hmin, count);
    dim3 grid(Cout[l] / 16, GY);
    if (l == 0)
      k_edge_wmma<1><<<grid, 256, 0, stream>>>(fin, stride, incol[l], Chalf[l], Cin[l], Cout[l],
                                               We[l], idx, PS, PQ, hmax, hmin);
    else if (l == 3)
      k_edge_wmma<8><<<grid, 256, 0, stream>>>(fin, stride, incol[l], Chalf[l], Cin[l], Cout[l],
                                               We[l], idx, PS, PQ, hmax, hmin);
    else
      k_edge_wmma<4><<<grid, 256, 0, stream>>>(fin, stride, incol[l], Chalf[l], Cin[l], Cout[l],
                                               We[l], idx, PS, PQ, hmax, hmin);
    k_stats_reduce<<<2, 256, 0, stream>>>(PS, PQ, ssum, ssq, Cout[l]);
    k_edge_apply<<<(count + 255) / 256, 256, 0, stream>>>(hmax, hmin, ssum, ssq, ge[l], be[l],
                                                          fcat, Cout[l], outcol[l], invcnt_e);
  }

  // 3) embedding GEMM + BN + leaky + global max/mean pooling
  k_emb_wmma<<<dim3(32, GY), 256, 0, stream>>>(fcat, Wemb, hemb, PS, PQ);
  k_stats_reduce<<<2, 256, 0, stream>>>(PS, PQ, ssum, ssq, 512);
  k_emb_pool<<<16, 256, 0, stream>>>(hemb, ssum, ssq, gemb, bemb, pooled);

  // 4) MLP head -> (8,40)
  k_head<<<1, 512, 0, stream>>>(pooled, Wl1, gl1, bl1, Wl2, gl2, bl2, Wf, bf, out);
}
